// bilstm_crf_22892175688470
// MI455X (gfx1250) — matrix operand order repaired and checked by reading, not compiled, not run on hardware
//
#include <hip/hip_runtime.h>
#include <hip/hip_bf16.h>
#include <cstdint>

// Problem constants (from reference): VOCAB=32000, E=256, H=512, TAGS=32, T=8192
#define T_SEQ 8192
#define E_DIM 256
#define H_DIM 512
#define G4H   2048   // 4*H
#define TAGS_N 32

typedef __bf16 bf16_t;
typedef __attribute__((ext_vector_type(16))) __bf16 v16bf;
typedef __attribute__((ext_vector_type(8)))  float  v8f;
typedef __attribute__((ext_vector_type(4)))  unsigned int v4u;
typedef __attribute__((ext_vector_type(8)))  int v8i;
typedef __attribute__((ext_vector_type(4)))  int v4i;

#if defined(__has_builtin)
#  if __has_builtin(__builtin_amdgcn_tensor_load_to_lds)
#    define HAVE_TDM 1
#  endif
#endif

// ---------------------------------------------------------------------------
// Prep kernels
// ---------------------------------------------------------------------------
__global__ void gather_cast_kernel(const int* __restrict__ sent,
                                   const float* __restrict__ emb,
                                   bf16_t* __restrict__ xbf) {
  int i = blockIdx.x * 256 + threadIdx.x;        // T*E threads exactly
  int t = i >> 8;                                // / E_DIM (256)
  int e = i & (E_DIM - 1);
  xbf[i] = (bf16_t)emb[(size_t)sent[t] * E_DIM + e];
}

__global__ void cast_kernel(const float* __restrict__ src,
                            bf16_t* __restrict__ dst, int n) {
  int i = blockIdx.x * 256 + threadIdx.x;
  if (i < n) dst[i] = (bf16_t)src[i];
}

__global__ void bias_sum_kernel(const float* __restrict__ a,
                                const float* __restrict__ b,
                                float* __restrict__ o, int n) {
  int i = blockIdx.x * 256 + threadIdx.x;
  if (i < n) o[i] = a[i] + b[i];
}

__global__ void init_ws_kernel(unsigned* __restrict__ bar) {
  if (threadIdx.x < 8) bar[threadIdx.x] = 0u;    // grid-barrier counters/gens
}

// ---------------------------------------------------------------------------
// Tensor Data Mover: issue a 2D tile load (rows x 32 cols of bf16) into LDS.
// D# bitfields per CDNA5 ISA 8.3/8.4: group0 = {count, lds_addr, global_addr,
// type=2}; group1 = {data_size=2B, tensor_dim0=K, tensor_dim1=nrows_total,
// tile_dim0=32, tile_dim1=rows, tensor_dim0_stride=K}. Tracked by TENSORcnt.
// ---------------------------------------------------------------------------
#if HAVE_TDM
__device__ __forceinline__ void tdm_load_tile_b32cols(
    const bf16_t* gsrc, unsigned lds_off, unsigned rows,
    unsigned K, unsigned nrows_total) {
  unsigned long long ga = (unsigned long long)(uintptr_t)gsrc;
  v4u g0;
  g0[0] = 1u;                                        // count=1, user mode
  g0[1] = lds_off;                                   // lds_addr
  g0[2] = (unsigned)(ga & 0xFFFFFFFFu);              // global_addr[31:0]
  g0[3] = (unsigned)((ga >> 32) & 0x01FFFFFFu) | (2u << 30);  // addr[56:32], type=2
  v8i g1;
  g1[0] = (int)(1u << 16);                           // data_size=1 (2 bytes)
  g1[1] = (int)((K & 0xFFFFu) << 16);                // tensor_dim0[15:0]
  g1[2] = (int)(((K >> 16) & 0xFFFFu) | ((nrows_total & 0xFFFFu) << 16));
  g1[3] = (int)(((nrows_total >> 16) & 0xFFFFu) | (32u << 16)); // tile_dim0=32
  g1[4] = (int)(rows & 0xFFFFu);                     // tile_dim1, tile_dim2=0
  g1[5] = (int)K;                                    // tensor_dim0_stride[31:0]
  g1[6] = 0;                                         // stride0[47:32], stride1 lo
  g1[7] = 0;
  v4i gz = {0, 0, 0, 0};
#if __clang_major__ >= 23
  v8i gz8 = {0, 0, 0, 0, 0, 0, 0, 0};
  __builtin_amdgcn_tensor_load_to_lds(g0, g1, gz, gz, gz8, 0);
#else
  __builtin_amdgcn_tensor_load_to_lds(g0, g1, gz, gz, 0);
#endif
}
#endif

// ---------------------------------------------------------------------------
// WMMA bf16 GEMM:  C(MxN, f32) = A(MxK, bf16) @ B(NxK, bf16)^T + bias(N)
// 256 threads = 8 wave32 waves; block tile 64(M) x BN(N); K step 32.
// A is staged into LDS pre-swizzled into a contiguous-k (k = 16*hi + s) per-lane fragment
// order so each lane reads its whole 32B fragment contiguously (ds_load_b128).
// B tiles are fetched by the Tensor Data Mover (wave 0 issues, waits on
// TENSORcnt, block barrier publishes) with a manual-copy fallback.
// ---------------------------------------------------------------------------
template <int BN>
__global__ __launch_bounds__(256)
void wmma_gemm_bias_kernel(const bf16_t* __restrict__ A,
                           const bf16_t* __restrict__ B,
                           const float*  __restrict__ bias,
                           float* __restrict__ C,
                           int M, int N, int K) {
  constexpr int NT = BN / 32;            // 16x16 n-tiles per wave (1 or 2)
  __shared__ bf16_t Asf[4][32][16];      // [mtile][fragLane][elem], swizzled
  __shared__ bf16_t Bs[BN][32];          // [n][k] row-major tile

  const int lane = threadIdx.x & 31;
  const int wv   = threadIdx.x >> 5;     // 0..7
  const int tm   = wv & 3;               // M tile (0..3)
  const int tn   = wv >> 2;              // N strip (0..1)
  const int hi   = lane >> 4;
  const int l15  = lane & 15;
  const int bm   = blockIdx.x * 64;
  const int bn   = blockIdx.y * BN;

  // A cooperative load: each thread moves 8 contiguous bf16 (16B).
  // Fragment swizzle: element (m, k) lives in lane (m&15)+16*(k>>4) at
  // slot k&15; an 8-aligned K-octet is contiguous in one fragment.
  const int aIdx = threadIdx.x * 8;
  const int ar   = aIdx >> 5;            // row in tile, 0..63
  const int ac   = aIdx & 31;            // 0, 8, 16, 24
  const int amt  = ar >> 4;
  const int fragLane = (ar & 15) + (ac >> 4) * 16;
  const int fragOff  = ((ac >> 3) & 1) * 8;

  v8f acc[NT] = {};
  for (int k0 = 0; k0 < K; k0 += 32) {
    *reinterpret_cast<float4*>(&Asf[amt][fragLane][fragOff]) =
        *reinterpret_cast<const float4*>(A + (size_t)(bm + ar) * K + k0 + ac);
    __builtin_prefetch(A + (size_t)(bm + ar) * K + k0 + 32 + ac, 0, 1);
#if HAVE_TDM
    if (wv == 0) {
      tdm_load_tile_b32cols(B + (size_t)bn * K + k0,
                            (unsigned)(uintptr_t)&Bs[0][0],
                            (unsigned)BN, (unsigned)K, (unsigned)N);
      __builtin_amdgcn_s_wait_tensorcnt(0);
    }
#else
    {
      constexpr int BE = BN * 32 / 256;  // elems per thread: 4 or 8
      const int bIdx = threadIdx.x * BE;
      const int br = bIdx >> 5, bc = bIdx & 31;
      if constexpr (BE == 8)
        *reinterpret_cast<float4*>(&Bs[br][bc]) =
            *reinterpret_cast<const float4*>(B + (size_t)(bn + br) * K + k0 + bc);
      else
        *reinterpret_cast<float2*>(&Bs[br][bc]) =
            *reinterpret_cast<const float2*>(B + (size_t)(bn + br) * K + k0 + bc);
    }
#endif
    __syncthreads();

    v16bf a = *reinterpret_cast<const v16bf*>(&Asf[tm][lane][0]);
#pragma unroll
    for (int nt = 0; nt < NT; ++nt) {
      // B 32x16 layout: lane's 16 elems are contiguous within a Bs row
      v16bf b = *reinterpret_cast<const v16bf*>(&Bs[(tn * NT + nt) * 16 + l15][hi * 16]);
      acc[nt] = __builtin_amdgcn_wmma_f32_16x16x32_bf16(false, a, false, b,
                                                        (short)0, acc[nt],
                                                        false, false);
    }
    __syncthreads();
  }

  // C/D layout: VGPR r, lane L -> row r + (L>>4)*8, col L&15
#pragma unroll
  for (int nt = 0; nt < NT; ++nt) {
    int n = bn + (tn * NT + nt) * 16 + l15;
    float bval = bias[n];
    float* cp = C + (size_t)(bm + tm * 16 + hi * 8) * N + n;
#pragma unroll
    for (int r = 0; r < 8; ++r) cp[(size_t)r * N] = acc[nt][r] + bval;
  }
}

// ---------------------------------------------------------------------------
// Persistent BiLSTM scan. 32 blocks x 128 threads.
//   blocks 0..15  : forward direction,  block b owns units [b*32, b*32+32)
//   blocks 16..31 : backward direction (iterates t = T-1..0)
// Each block keeps its 128x512 fp32 W_hh slice resident in LDS (row stride
// 514 for alignment + bank spread; ~257KB, needs CDNA5's 320KB WGP LDS).
// h is exchanged through L2 with device-scope atomics + a monotonic-
// generation global barrier; h is ping-pong buffered by step parity.
// ---------------------------------------------------------------------------
#define WPAD 514
#define SCAN_LDS_BYTES ((128 * WPAD + H_DIM + 128) * 4)

__device__ inline void grid_bar(unsigned* cnt, unsigned* gen, unsigned target) {
  __syncthreads();
  if (threadIdx.x == 0) {
    __threadfence();
    unsigned v = atomicAdd(cnt, 1u) + 1u;
    if (v == target * 16u) {
      __hip_atomic_store(gen, target, __ATOMIC_RELEASE, __HIP_MEMORY_SCOPE_AGENT);
    } else {
      while (__hip_atomic_load(gen, __ATOMIC_ACQUIRE, __HIP_MEMORY_SCOPE_AGENT) < target)
        __builtin_amdgcn_s_sleep(1);
    }
    __threadfence();
  }
  __syncthreads();
}

__global__ __launch_bounds__(128)
void bilstm_scan_kernel(const float* __restrict__ pre_f,
                        const float* __restrict__ pre_b,
                        const float* __restrict__ Whh_f,
                        const float* __restrict__ Whh_b,
                        const float* __restrict__ h0,
                        const float* __restrict__ c0,
                        float* __restrict__ hbuf,    // [2 slots][2 dirs][H]
                        bf16_t* __restrict__ hcat,   // [T][2H] bf16
                        unsigned* __restrict__ bar)  // [4]: cnt_f,cnt_b,gen_f,gen_b
{
  extern __shared__ float smem[];
  float* Wl = smem;                 // 128 x WPAD
  float* hl = smem + 128 * WPAD;    // H
  float* gv = hl + H_DIM;           // 128 gate pre-activations

  // Cluster-aware path: if launched as a workgroup cluster, also rendezvous
  // on the hardware cluster barrier (NOP when ClusterID==0 per ISA 3.2).
  if (__builtin_amdgcn_cluster_id_x() != 0) __builtin_amdgcn_s_cluster_barrier();

  const int dir = blockIdx.x >> 4;
  const int blk = blockIdx.x & 15;
  const int bu  = blk * 32;
  const int tid = threadIdx.x;
  const float* pre = dir ? pre_b : pre_f;
  const float* Whh = dir ? Whh_b : Whh_f;
  unsigned* cnt = bar + dir;
  unsigned* gen = bar + 2 + dir;

  // Stage this block's W_hh slice: LDS row r <-> global gate-row (r/32)*H + bu + r%32
  for (int j = tid; j < 128 * H_DIM; j += 128) {
    int r = j >> 9, k = j & (H_DIM - 1);
    int grow = (r >> 5) * H_DIM + bu + (r & 31);
    Wl[r * WPAD + k] = Whh[(size_t)grow * H_DIM + k];
  }

  float c = 0.f;
  if (tid < 32) {
    c = c0[dir * H_DIM + bu + tid];
    __hip_atomic_store(&hbuf[(0 * 2 + dir) * H_DIM + bu + tid],
                       h0[dir * H_DIM + bu + tid],
                       __ATOMIC_RELAXED, __HIP_MEMORY_SCOPE_AGENT);
  }
  grid_bar(cnt, gen, 1u);

  for (int t = 0; t < T_SEQ; ++t) {
    const int ts = dir ? (T_SEQ - 1 - t) : t;
    const int rd = t & 1;           // read slot
    const int wr = rd ^ 1;          // write slot

    // broadcast-stage h_{t-1} into LDS
#pragma unroll
    for (int i = 0; i < 4; ++i) {
      int k = tid + i * 128;
      hl[k] = __hip_atomic_load(&hbuf[(rd * 2 + dir) * H_DIM + k],
                                __ATOMIC_RELAXED, __HIP_MEMORY_SCOPE_AGENT);
    }
    __syncthreads();

    // one gate row per thread: 512-MAC dot from LDS-resident weights
    const float2* wrow = reinterpret_cast<const float2*>(Wl + (size_t)tid * WPAD);
    const float2* h2   = reinterpret_cast<const float2*>(hl);
    float s = pre[(size_t)ts * G4H + (tid >> 5) * H_DIM + bu + (tid & 31)];
#pragma unroll 8
    for (int k = 0; k < H_DIM / 2; ++k) {
      float2 w = wrow[k], hh = h2[k];
      s = __builtin_fmaf(w.x, hh.x, s);
      s = __builtin_fmaf(w.y, hh.y, s);
    }
    gv[tid] = s;
    __syncthreads();

    if (tid < 32) {
      float ig = 1.f / (1.f + __expf(-gv[tid]));
      float fg = 1.f / (1.f + __expf(-gv[32 + tid]));
      float gg = tanhf(gv[64 + tid]);
      float og = 1.f / (1.f + __expf(-gv[96 + tid]));
      c = fg * c + ig * gg;
      float h = og * tanhf(c);
      __hip_atomic_store(&hbuf[(wr * 2 + dir) * H_DIM + bu + tid], h,
                         __ATOMIC_RELAXED, __HIP_MEMORY_SCOPE_AGENT);
      hcat[(size_t)ts * (2 * H_DIM) + dir * H_DIM + bu + tid] = (bf16_t)h;
    }
    grid_bar(cnt, gen, (unsigned)(t + 2));
  }
}

// ---------------------------------------------------------------------------
extern "C" void kernel_launch(void* const* d_in, const int* in_sizes, int n_in,
                              void* d_out, int out_size, void* d_ws, size_t ws_size,
                              hipStream_t stream) {
  const int*   sent  = (const int*)  d_in[0];
  const float* emb   = (const float*)d_in[1];
  const float* Wih_f = (const float*)d_in[2];
  const float* Whh_f = (const float*)d_in[3];
  const float* bih_f = (const float*)d_in[4];
  const float* bhh_f = (const float*)d_in[5];
  const float* Wih_b = (const float*)d_in[6];
  const float* Whh_b = (const float*)d_in[7];
  const float* bih_b = (const float*)d_in[8];
  const float* bhh_b = (const float*)d_in[9];
  const float* Wout  = (const float*)d_in[10];
  const float* bout  = (const float*)d_in[11];
  const float* h0    = (const float*)d_in[12];
  const float* c0    = (const float*)d_in[13];
  float* out = (float*)d_out;   // (T, TAGS) fp32

  // workspace carve-out
  char* ws = (char*)d_ws;
  size_t off = 0;
  auto alloc = [&](size_t bytes) -> void* {
    void* p = ws + off;
    off = (off + bytes + 255) & ~(size_t)255;
    return p;
  };
  bf16_t*   xbf   = (bf16_t*)alloc((size_t)T_SEQ * E_DIM * sizeof(bf16_t));
  bf16_t*   wfbf  = (bf16_t*)alloc((size_t)G4H * E_DIM * sizeof(bf16_t));
  bf16_t*   wbbf  = (bf16_t*)alloc((size_t)G4H * E_DIM * sizeof(bf16_t));
  bf16_t*   wobf  = (bf16_t*)alloc((size_t)TAGS_N * 2 * H_DIM * sizeof(bf16_t));
  float*    bsf   = (float*) alloc((size_t)G4H * sizeof(float));
  float*    bsb   = (float*) alloc((size_t)G4H * sizeof(float));
  float*    pre_f = (float*) alloc((size_t)T_SEQ * G4H * sizeof(float));
  float*    pre_b = (float*) alloc((size_t)T_SEQ * G4H * sizeof(float));
  bf16_t*   hcat  = (bf16_t*)alloc((size_t)T_SEQ * 2 * H_DIM * sizeof(bf16_t));
  float*    hbuf  = (float*) alloc((size_t)2 * 2 * H_DIM * sizeof(float));
  unsigned* bar   = (unsigned*)alloc(8 * sizeof(unsigned));

  // phase 0: gather/cast/bias/init
  gather_cast_kernel<<<(T_SEQ * E_DIM) / 256, 256, 0, stream>>>(sent, emb, xbf);
  cast_kernel<<<(G4H * E_DIM + 255) / 256, 256, 0, stream>>>(Wih_f, wfbf, G4H * E_DIM);
  cast_kernel<<<(G4H * E_DIM + 255) / 256, 256, 0, stream>>>(Wih_b, wbbf, G4H * E_DIM);
  cast_kernel<<<(TAGS_N * 2 * H_DIM + 255) / 256, 256, 0, stream>>>(Wout, wobf, TAGS_N * 2 * H_DIM);
  bias_sum_kernel<<<(G4H + 255) / 256, 256, 0, stream>>>(bih_f, bhh_f, bsf, G4H);
  bias_sum_kernel<<<(G4H + 255) / 256, 256, 0, stream>>>(bih_b, bhh_b, bsb, G4H);
  init_ws_kernel<<<1, 32, 0, stream>>>(bar);

  // phase 1: input GEMMs (WMMA bf16 + TDM B-tile loads)
  wmma_gemm_bias_kernel<64><<<dim3(T_SEQ / 64, G4H / 64), 256, 0, stream>>>(
      xbf, wfbf, bsf, pre_f, T_SEQ, G4H, E_DIM);
  wmma_gemm_bias_kernel<64><<<dim3(T_SEQ / 64, G4H / 64), 256, 0, stream>>>(
      xbf, wbbf, bsb, pre_b, T_SEQ, G4H, E_DIM);

  // phase 2: persistent bidirectional scan (LDS-resident W_hh, grid barrier)
  hipFuncSetAttribute((const void*)bilstm_scan_kernel,
                      hipFuncAttributeMaxDynamicSharedMemorySize, SCAN_LDS_BYTES);
  bilstm_scan_kernel<<<32, 128, SCAN_LDS_BYTES, stream>>>(
      pre_f, pre_b, Whh_f, Whh_b, h0, c0, hbuf, hcat, bar);

  // phase 3: output GEMM (WMMA bf16), out = hcat @ W_out^T + b_out
  wmma_gemm_bias_kernel<32><<<dim3(T_SEQ / 64, TAGS_N / 32), 256, 0, stream>>>(
      hcat, wobf, bout, out, T_SEQ, TAGS_N, 2 * H_DIM);
}
